// AttentionHead_51178830299302
// MI455X (gfx1250) — matrix operand order repaired and checked by reading, not compiled, not run on hardware
//
#include <hip/hip_runtime.h>
#include <hip/hip_bf16.h>

// ---------- types ----------
typedef __bf16 bf16_t;
typedef __attribute__((ext_vector_type(16))) __bf16 v16bf;
typedef __attribute__((ext_vector_type(8)))  __bf16 v8bf;
typedef __attribute__((ext_vector_type(8)))  float  v8f;
typedef int v4i_vs __attribute__((vector_size(16)));   // matches builtin param type

#define NEG_VAL (-100000.0f)

// ---------- async global->LDS copy (CDNA5 ASYNCcnt path), with fallback ----------
#if defined(__has_builtin)
#if __has_builtin(__builtin_amdgcn_global_load_async_to_lds_b128)
#define HAVE_ASYNC_LDS 1
#endif
#endif
#ifndef HAVE_ASYNC_LDS
#define HAVE_ASYNC_LDS 0
#endif

// Copy 16 bytes per lane from global to LDS.
__device__ __forceinline__ void async_copy16(const bf16_t* gsrc, bf16_t* ldst) {
#if HAVE_ASYNC_LDS
  __builtin_amdgcn_global_load_async_to_lds_b128(
      (v4i_vs*)gsrc, (v4i_vs*)ldst, /*offset=*/0, /*cpol=*/0);
#else
  *reinterpret_cast<v8bf*>(ldst) = *reinterpret_cast<const v8bf*>(gsrc);
#endif
}

__device__ __forceinline__ void async_wait0() {
#if HAVE_ASYNC_LDS
#if __has_builtin(__builtin_amdgcn_s_wait_asynccnt)
  __builtin_amdgcn_s_wait_asynccnt(0);
#else
  asm volatile("s_wait_asynccnt 0x0" ::: "memory");
#endif
#endif
}

// Build a v16bf A/B operand from two contiguous 8-element (16B) chunks.
__device__ __forceinline__ v16bf ld16(const bf16_t* p0, const bf16_t* p1) {
  v16bf r;
  v8bf lo = *reinterpret_cast<const v8bf*>(p0);
  v8bf hi = *reinterpret_cast<const v8bf*>(p1);
#pragma unroll
  for (int i = 0; i < 8; ++i) { r[i] = lo[i]; r[i + 8] = hi[i]; }
  return r;
}

__device__ __forceinline__ v8f wmma_bf16(v16bf a, v16bf b, v8f c) {
  return __builtin_amdgcn_wmma_f32_16x16x32_bf16(
      /*neg_a=*/false, a, /*neg_b=*/false, b,
      /*c_mod=*/(short)0, c, /*reuse_a=*/false, /*reuse_b=*/false);
}

// ---------- stage 0: fp32 -> bf16 conversions ----------
__global__ void cvt_bf16_kernel(const float* __restrict__ in,
                                bf16_t* __restrict__ out, long n) {
  long i = (long)blockIdx.x * blockDim.x + threadIdx.x;
  long stride = (long)gridDim.x * blockDim.x;
  for (; i < n; i += stride) out[i] = (bf16_t)in[i];
}

// wT[h*E + e] = (bf16) w[e*H + h]
__global__ void transpose_w_kernel(const float* __restrict__ w,
                                   bf16_t* __restrict__ wT, int E, int H) {
  long i = (long)blockIdx.x * blockDim.x + threadIdx.x;
  long n = (long)E * H;
  if (i >= n) return;
  int e = (int)(i % E);
  int h = (int)(i / E);
  wT[(long)h * E + e] = (bf16_t)w[(long)e * H + h];
}

// ---------- stage 1: QKV projection ----------
// Block = 4 waves, each wave owns one 16-row tile. Weight tiles for each
// K-step (3 matrices x 64 h x 32 e bf16 = 12KB) are async-staged into LDS
// once per block and double-buffered.
__global__ __launch_bounds__(128, 1)
void qkv_proj_kernel(const bf16_t* __restrict__ hid,
                     const bf16_t* __restrict__ wqT,
                     const bf16_t* __restrict__ wkT,
                     const bf16_t* __restrict__ wvT,
                     const float* __restrict__ bq,
                     const float* __restrict__ bk,
                     const float* __restrict__ bv,
                     bf16_t* __restrict__ qo,   // [B*S, H]
                     bf16_t* __restrict__ ko,   // [B*S, H]
                     bf16_t* __restrict__ vTo,  // [B, H, S]
                     int S, int E, int H) {
  __shared__ alignas(16) bf16_t wslab[2][3 * 64 * 32];  // [m][h][e_local]

  const int tid  = threadIdx.x;
  const int wave = (blockIdx.x * blockDim.x + tid) >> 5;
  const int lane = tid & 31;
  const int half = lane >> 4;
  const int n    = lane & 15;
  const long row0 = (long)wave * 16;

  auto fill = [&](int buf, int e0) {
#pragma unroll
    for (int c = 0; c < 6; ++c) {               // 768 x 16B chunks / 128 thr
      const int idx  = tid + c * 128;           // 0..767
      const int elem = idx * 8;                 // bf16 element offset
      const int m    = elem >> 11;              // / (64*32)
      const int rem  = elem & 2047;
      const int h    = rem >> 5;
      const int e    = rem & 31;
      const bf16_t* src =
          ((m == 0) ? wqT : (m == 1) ? wkT : wvT) + (long)h * E + e0 + e;
      async_copy16(src, &wslab[buf][elem]);
    }
  };

  v8f accq[4], acck[4], accv[4];
#pragma unroll
  for (int t = 0; t < 4; ++t)
#pragma unroll
    for (int r = 0; r < 8; ++r) { accq[t][r] = 0.f; acck[t][r] = 0.f; accv[t][r] = 0.f; }

  fill(0, 0);
  const int iters = E / 32;
  for (int it = 0; it < iters; ++it) {
    const int e0  = it * 32;
    const int cur = it & 1;
    async_wait0();
    __syncthreads();
    if (it + 1 < iters) fill(cur ^ 1, e0 + 32);  // prefetch next K-step

    const bf16_t* arow = hid + (row0 + n) * (long)E + e0;
    v16bf a = ld16(arow + half * 16, arow + half * 16 + 8);
#pragma unroll
    for (int nt = 0; nt < 4; ++nt) {
      const int h = nt * 16 + n;
      const bf16_t* pq = &wslab[cur][(0 * 64 + h) * 32 + half * 16];
      const bf16_t* pk = &wslab[cur][(1 * 64 + h) * 32 + half * 16];
      const bf16_t* pv = &wslab[cur][(2 * 64 + h) * 32 + half * 16];
      accq[nt] = wmma_bf16(a, ld16(pq, pq + 8), accq[nt]);
      acck[nt] = wmma_bf16(a, ld16(pk, pk + 8), acck[nt]);
      accv[nt] = wmma_bf16(a, ld16(pv, pv + 8), accv[nt]);
    }
  }

  // Epilogue: bias, store q/k row-major bf16 and v transposed.
#pragma unroll
  for (int nt = 0; nt < 4; ++nt) {
    const int h = nt * 16 + n;
    const float biq = bq[h], bik = bk[h], biv = bv[h];
#pragma unroll
    for (int r = 0; r < 8; ++r) {
      const long srow = row0 + r + 8 * half;
      qo[srow * H + h] = (bf16_t)(accq[nt][r] + biq);
      ko[srow * H + h] = (bf16_t)(acck[nt][r] + bik);
      const int bidx = (int)(srow / S);
      const int sl   = (int)(srow % S);
      vTo[((long)bidx * H + h) * S + sl] = (bf16_t)(accv[nt][r] + biv);
    }
  }
}

// ---------- stage 2: flash attention ----------
// Block = 4 waves, all in the same batch (S/16 = 128 tiles per batch, block
// covers 4 consecutive tiles). K (32x64) and V^T (64x32) blocks are async-
// staged into LDS, shared by all 4 waves, double-buffered.
__global__ __launch_bounds__(128, 1)
void attn_kernel(const bf16_t* __restrict__ q,   // [B*S, H]
                 const bf16_t* __restrict__ k,   // [B*S, H]
                 const bf16_t* __restrict__ vT,  // [B, H, S]
                 const int* __restrict__ amask,  // [B, S]
                 float* __restrict__ out,        // [B*S, H]
                 int S, int H) {
  __shared__ alignas(16) bf16_t kslab[2][32 * 64];  // [key][h]
  __shared__ alignas(16) bf16_t vslab[2][64 * 32];  // [h][key]
  __shared__ alignas(16) bf16_t Pl[4][16][32];      // per-wave P staging

  const int tid  = threadIdx.x;
  const int wib  = tid >> 5;
  const int wave = blockIdx.x * 4 + wib;
  const int lane = tid & 31;
  const int half = lane >> 4;
  const int n    = lane & 15;

  const long qrow0 = (long)wave * 16;
  const int  b  = (int)(qrow0 / S);   // same for all 4 waves in block
  const int  q0 = (int)(qrow0 % S);

  const bf16_t* kbase_g = k + (long)b * S * H;   // [S][H]
  const bf16_t* vbase_g = vT + (long)b * H * S;  // [H][S]

  auto fill = [&](int buf, int kb) {
#pragma unroll
    for (int c = 0; c < 2; ++c) {                // 256 x 16B chunks / 128 thr
      const int idx  = tid + c * 128;            // 0..255
      const int elem = idx * 8;                  // 0..2047
      // K slab: contiguous 4KB ([key][h] matches memory [S][H] slice)
      async_copy16(kbase_g + (long)kb * H + elem, &kslab[buf][elem]);
      // V slab: row h of V^T, 32 keys
      const int h = elem >> 5;
      const int e = elem & 31;
      async_copy16(vbase_g + (long)h * S + kb + e, &vslab[buf][elem]);
    }
  };

  // Q as A operand: 16x64 -> two 16x32 K-steps, kept in registers.
  const bf16_t* qrow = q + (qrow0 + n) * (long)H;
  v16bf aq0 = ld16(qrow + half * 16,      qrow + half * 16 + 8);
  v16bf aq1 = ld16(qrow + 32 + half * 16, qrow + 32 + half * 16 + 8);

  // Query-row mask bits (uniform via ballot).
  int qm = (lane < 16) ? amask[(long)b * S + q0 + lane] : 0;
  const unsigned qbits = (unsigned)(__ballot(qm != 0) & 0xFFFFull);

  v8f acc[4];
  float mrun[8], lrun[8];
#pragma unroll
  for (int t = 0; t < 4; ++t)
#pragma unroll
    for (int r = 0; r < 8; ++r) acc[t][r] = 0.f;
#pragma unroll
  for (int r = 0; r < 8; ++r) { mrun[r] = -1e30f; lrun[r] = 0.f; }

  fill(0, 0);
  int it = 0;
  for (int kb = 0; kb < S; kb += 32, ++it) {
    const int cur = it & 1;
    async_wait0();
    __syncthreads();
    if (kb + 32 < S) fill(cur ^ 1, kb + 32);     // prefetch next key block

    const unsigned kbits = (unsigned)__ballot(amask[(long)b * S + kb + lane] != 0);

    // Two 16x16 score tiles.
    v8f st[2];
#pragma unroll
    for (int t = 0; t < 2; ++t) {
      const bf16_t* krow = &kslab[cur][(t * 16 + n) * 64];
      v16bf bk0 = ld16(krow + half * 16,      krow + half * 16 + 8);
      v16bf bk1 = ld16(krow + 32 + half * 16, krow + 32 + half * 16 + 8);
      v8f s;
#pragma unroll
      for (int r = 0; r < 8; ++r) s[r] = 0.f;
      s = wmma_bf16(aq0, bk0, s);
      s = wmma_bf16(aq1, bk1, s);
#pragma unroll
      for (int r = 0; r < 8; ++r) {
        const int qi = q0 + r + 8 * half;
        const int ki = kb + t * 16 + n;
        const int mq = (qbits >> (r + 8 * half)) & 1;
        const int mk = (kbits >> (t * 16 + n)) & 1;
        const bool allowed = (mq && mk && (ki <= qi)) || (!mq && !mk);
        s[r] = s[r] * 0.125f + (allowed ? 0.f : NEG_VAL);
      }
      st[t] = s;
    }

    // Online softmax (row reductions across the 16 lanes of each half).
#pragma unroll
    for (int r = 0; r < 8; ++r) {
      float rm = fmaxf(st[0][r], st[1][r]);
      rm = fmaxf(rm, __shfl_xor(rm, 1, 32));
      rm = fmaxf(rm, __shfl_xor(rm, 2, 32));
      rm = fmaxf(rm, __shfl_xor(rm, 4, 32));
      rm = fmaxf(rm, __shfl_xor(rm, 8, 32));
      const float mnew  = fmaxf(mrun[r], rm);
      const float alpha = __expf(mrun[r] - mnew);
      mrun[r] = mnew;
      const float p0 = __expf(st[0][r] - mnew);
      const float p1 = __expf(st[1][r] - mnew);
      Pl[wib][r + 8 * half][n]      = (bf16_t)p0;
      Pl[wib][r + 8 * half][16 + n] = (bf16_t)p1;
      float ps = p0 + p1;
      ps += __shfl_xor(ps, 1, 32);
      ps += __shfl_xor(ps, 2, 32);
      ps += __shfl_xor(ps, 4, 32);
      ps += __shfl_xor(ps, 8, 32);
      lrun[r] = lrun[r] * alpha + ps;
#pragma unroll
      for (int nt = 0; nt < 4; ++nt) acc[nt][r] *= alpha;
    }

    // Re-read P in A-operand layout (same-wave LDS ops are in-order).
    const bf16_t* pr = &Pl[wib][n][0];
    v16bf pa = ld16(pr + half * 16, pr + half * 16 + 8);

    // O += P(16x32) * V(32x64), B operand from the V^T LDS slab.
#pragma unroll
    for (int nt = 0; nt < 4; ++nt) {
      const bf16_t* vrow = &vslab[cur][(nt * 16 + n) * 32 + half * 16];
      acc[nt] = wmma_bf16(pa, ld16(vrow, vrow + 8), acc[nt]);
    }
  }

  // Normalize (8 reciprocals) and store fp32 output.
#pragma unroll
  for (int r = 0; r < 8; ++r) {
    const float inv = 1.0f / lrun[r];
    const long orow = (qrow0 + r + 8 * half) * H;
#pragma unroll
    for (int nt = 0; nt < 4; ++nt) {
      out[orow + nt * 16 + n] = acc[nt][r] * inv;
    }
  }
}

// ---------- host launch ----------
extern "C" void kernel_launch(void* const* d_in, const int* in_sizes, int n_in,
                              void* d_out, int out_size, void* d_ws, size_t ws_size,
                              hipStream_t stream) {
  const int B = 8, S = 2048, E = 1024, H = 64;

  const float* hidden = (const float*)d_in[0];
  const int*   amask  = (const int*)d_in[1];
  const float* Wq = (const float*)d_in[2];
  const float* bq = (const float*)d_in[3];
  const float* Wk = (const float*)d_in[4];
  const float* bk = (const float*)d_in[5];
  const float* Wv = (const float*)d_in[6];
  const float* bv = (const float*)d_in[7];
  float* out = (float*)d_out;

  char* base = (char*)d_ws;
  size_t off = 0;
  auto carve = [&](size_t bytes) -> char* {
    char* p = base + off;
    off = (off + bytes + 255) & ~(size_t)255;
    return p;
  };
  bf16_t* hidb = (bf16_t*)carve((size_t)B * S * E * sizeof(bf16_t));
  bf16_t* wqT  = (bf16_t*)carve((size_t)H * E * sizeof(bf16_t));
  bf16_t* wkT  = (bf16_t*)carve((size_t)H * E * sizeof(bf16_t));
  bf16_t* wvT  = (bf16_t*)carve((size_t)H * E * sizeof(bf16_t));
  bf16_t* qb   = (bf16_t*)carve((size_t)B * S * H * sizeof(bf16_t));
  bf16_t* kbuf = (bf16_t*)carve((size_t)B * S * H * sizeof(bf16_t));
  bf16_t* vTb  = (bf16_t*)carve((size_t)B * H * S * sizeof(bf16_t));
  if (off > ws_size) return;

  cvt_bf16_kernel<<<2048, 256, 0, stream>>>(hidden, hidb, (long)B * S * E);
  const int tw_blocks = (E * H + 255) / 256;
  transpose_w_kernel<<<tw_blocks, 256, 0, stream>>>(Wq, wqT, E, H);
  transpose_w_kernel<<<tw_blocks, 256, 0, stream>>>(Wk, wkT, E, H);
  transpose_w_kernel<<<tw_blocks, 256, 0, stream>>>(Wv, wvT, E, H);

  const int tiles  = (B * S) / 16;  // 1024
  const int blocks = tiles / 4;     // 256
  qkv_proj_kernel<<<blocks, 128, 0, stream>>>(hidb, wqT, wkT, wvT, bq, bk, bv,
                                              qb, kbuf, vTb, S, E, H);
  attn_kernel<<<blocks, 128, 0, stream>>>(qb, kbuf, vTb, amask, out, S, H);
}
